// MultiHeadAttention_73160472920449
// MI455X (gfx1250) — compile-verified, matrix operand order checked by reading, not run on hardware
//
#include <hip/hip_runtime.h>
#include <hip/hip_bf16.h>
#include <stdint.h>

#define D_MODEL 512
#define N_HEADS 8
#define D_K     64
#define BATCH   16
#define SEQ     1024
#define M_TOTAL (BATCH * SEQ)   // 16384

typedef __attribute__((ext_vector_type(16))) __bf16 bf16x16;
typedef __attribute__((ext_vector_type(8)))  float  f32x8;

// ---------- helpers ----------

__device__ __forceinline__ unsigned short f2bf(float f) {
    union { float f; uint32_t u; } x; x.f = f;
    uint32_t r = (x.u + 0x7FFFu + ((x.u >> 16) & 1u)) >> 16;  // RNE
    return (unsigned short)r;
}

__device__ __forceinline__ f32x8 wmma_bf16(bf16x16 a, bf16x16 b, f32x8 c) {
    return __builtin_amdgcn_wmma_f32_16x16x32_bf16(
        /*neg_a=*/false, a, /*neg_b=*/false, b,
        /*c_mod=*/(short)0, c, /*reuse_a=*/false, /*reuse_b=*/false);
}

// WMMA 16-bit fragment swizzle (ISA 7.12.2): within a 32-wide K-chunk,
// element k lives in lane half hi=(k>>3)&1 (lanes 0-15 hold K{0..7,16..23},
// lanes 16-31 hold K{8..15,24..31}) at slot j.
__device__ __forceinline__ void kswz(int kk, int& hi, int& j) {
    hi = (kk >> 3) & 1;
    int klo = kk - 8 * hi;                 // {0..7} U {16..23}
    j = (klo < 8) ? klo : klo - 8;         // 0..15
}

// A-style fragment image: tiles of 16 rows, K-chunks of 32.
// idx = ((rowTile*16 + kChunk)*32 + lane)*16 + j   (lane = hi<<4 | row%16)
__device__ __forceinline__ size_t afrag_idx(int row, int k) {
    int hi, j; kswz(k & 31, hi, j);
    return (((size_t)(row >> 4) * 16 + (k >> 5)) * 32 + ((hi << 4) | (row & 15))) * 16 + j;
}

// Row-fragment layout per (b,h) for Q/K (B-operand of QK^T == row gather of K):
// per bh: 64 tiles of 16 rows, 2 chunks of 32 d, 32 lanes, 16 elems.
__device__ __forceinline__ size_t rowfrag_idx(size_t bh, int row, int d) {
    int hi, j; kswz(d & 31, hi, j);
    return ((bh * 64 + (row >> 4)) * 2 + (d >> 5)) * 512
         + (size_t)((((hi << 4) | (row & 15)) << 4) | j);
}

// Col-fragment layout per (b,h) for V (B-operand of P@V):
// per bh: 32 key-chunks of 32, 4 d-tiles of 16, 32 lanes, 16 elems.
__device__ __forceinline__ size_t colfrag_idx(size_t bh, int key, int d) {
    int hi, j; kswz(key & 31, hi, j);
    return ((bh * 32 + (key >> 5)) * 4 + (d >> 4)) * 512
         + (size_t)((((hi << 4) | (d & 15)) << 4) | j);
}

// ---------- conversion kernels: f32 -> bf16, pre-swizzled ----------

__global__ void k_conv_A(const float* __restrict__ src,
                         unsigned short* __restrict__ dst, int nElem) {
    int i = blockIdx.x * blockDim.x + threadIdx.x;       // over rows*512
    if (i >= nElem) return;
    int m = i >> 9, k = i & 511;
    dst[afrag_idx(m, k)] = f2bf(src[i]);
}

__global__ void k_conv_W(const float* __restrict__ src,
                         unsigned short* __restrict__ dst) {
    int i = blockIdx.x * blockDim.x + threadIdx.x;       // over 512*512, (k,n)
    int k = i >> 9, n = i & 511;
    int hi, j; kswz(k & 31, hi, j);
    size_t idx = (((size_t)(n >> 4) * 16 + (k >> 5)) * 32 + ((hi << 4) | (n & 15))) * 16 + j;
    dst[idx] = f2bf(src[i]);
}

// ---------- GEMM: C = A(Mx512 bf16 afrag) * W(512x512 bf16 bfrag) + bias ----------
// 256 thr = 8 waves; wave = 2x2 register-blocked 32x32 tile; block = 128(M)x64(N).
// No LDS, no barriers: 4 fragment loads (8x global_load_b128) feed 4 WMMAs.
// mode 0: f32 row-major out.  mode 1: bf16 rowfrag (Q/K).  mode 2: bf16 colfrag (V).

__global__ __launch_bounds__(256)
void k_gemm512(const unsigned short* __restrict__ Af,
               const unsigned short* __restrict__ Wf,
               const float* __restrict__ bias,
               float* __restrict__ outF,
               unsigned short* __restrict__ outHS,
               int mode) {
    const int t    = threadIdx.x;
    const int lane = t & 31;
    const int wv   = t >> 5;
    const int mt0  = blockIdx.x * 8 + (wv & 3) * 2;   // 16-row tile index
    const int nt0  = blockIdx.y * 4 + (wv >> 2) * 2;  // 16-col tile index

    f32x8 acc[2][2];
    #pragma unroll
    for (int mi = 0; mi < 2; ++mi)
        #pragma unroll
        for (int ni = 0; ni < 2; ++ni)
            acc[mi][ni] = (f32x8){0.f,0.f,0.f,0.f,0.f,0.f,0.f,0.f};

    for (int kc = 0; kc < 16; ++kc) {
        const unsigned short* a0p = Af + (((size_t)mt0 * 16 + kc) * 32 + lane) * 16;
        const unsigned short* a1p = a0p + 16 * 512;           // next m-tile
        const unsigned short* b0p = Wf + (((size_t)nt0 * 16 + kc) * 32 + lane) * 16;
        const unsigned short* b1p = b0p + 16 * 512;           // next n-tile
        if (kc < 15) {                                        // gfx1250 global_prefetch
            __builtin_prefetch(a0p + 512, 0, 1);
            __builtin_prefetch(b0p + 512, 0, 1);
        }
        bf16x16 a0 = *(const bf16x16*)a0p;
        bf16x16 a1 = *(const bf16x16*)a1p;
        bf16x16 b0 = *(const bf16x16*)b0p;
        bf16x16 b1 = *(const bf16x16*)b1p;
        acc[0][0] = wmma_bf16(a0, b0, acc[0][0]);
        acc[0][1] = wmma_bf16(a0, b1, acc[0][1]);
        acc[1][0] = wmma_bf16(a1, b0, acc[1][0]);
        acc[1][1] = wmma_bf16(a1, b1, acc[1][1]);
    }

    #pragma unroll
    for (int mi = 0; mi < 2; ++mi)
        #pragma unroll
        for (int ni = 0; ni < 2; ++ni)
            #pragma unroll
            for (int i = 0; i < 8; ++i) {
                int m = (mt0 + mi) * 16 + i + ((lane >> 4) << 3);
                int n = (nt0 + ni) * 16 + (lane & 15);
                float v = acc[mi][ni][i] + bias[n];
                if (mode == 0) {
                    outF[(size_t)m * 512 + n] = v;
                } else {
                    int b = m >> 10, s = m & 1023, h = n >> 6, d = n & 63;
                    size_t bh = (size_t)b * N_HEADS + h;
                    size_t idx = (mode == 1) ? rowfrag_idx(bh, s, d)
                                             : colfrag_idx(bh, s, d);
                    outHS[idx] = f2bf(v);
                }
            }
}

// ---------- attention: one block per (b, h, 16-query tile) ----------
// 128 thr = 4 waves. All Q/K/V fragments load straight from global (pre-swizzled,
// 32B/lane coalesced). LDS: scores 16x1024 f32 (64KB) + probs in P-fragment
// layout (32KB) + 1KB reduce scratch. Score and P@V loops are barrier-free.

__global__ __launch_bounds__(128)
void k_attention(const unsigned short* __restrict__ Qf,
                 const unsigned short* __restrict__ Kf,
                 const unsigned short* __restrict__ Vf,
                 float* __restrict__ attnW,
                 unsigned short* __restrict__ Cf) {
    extern __shared__ char smem[];
    float*          sS = (float*)smem;                          // 16x1024 f32
    unsigned short* sP = (unsigned short*)(smem + 65536);       // 32 chunks x 512
    float*          sR = (float*)(smem + 65536 + 32768);        // 2 x 16x8 f32

    const int t    = threadIdx.x;
    const int lane = t & 31;
    const int w    = t >> 5;                 // wave 0..3
    const int qt   = blockIdx.x;             // 16-query tile
    const int h    = blockIdx.y;
    const int b    = blockIdx.z;
    const size_t bh = (size_t)b * N_HEADS + h;

    // Q fragments (held in registers for the whole kernel)
    const unsigned short* Qb = Qf + (bh * 64 + qt) * 2 * 512;
    bf16x16 qa0 = *(const bf16x16*)(Qb + lane * 16);
    bf16x16 qa1 = *(const bf16x16*)(Qb + 512 + lane * 16);

    // ---- scores = scale * Q K^T ; wave w covers key tiles w, w+4, ... ----
    for (int g = 0; g < 16; ++g) {
        int kt = g * 4 + w;
        const unsigned short* Kb = Kf + (bh * 64 + kt) * 2 * 512;
        bf16x16 kb0 = *(const bf16x16*)(Kb + lane * 16);
        bf16x16 kb1 = *(const bf16x16*)(Kb + 512 + lane * 16);
        f32x8 acc = {0.f,0.f,0.f,0.f,0.f,0.f,0.f,0.f};
        acc = wmma_bf16(qa0, kb0, acc);
        acc = wmma_bf16(qa1, kb1, acc);
        #pragma unroll
        for (int i = 0; i < 8; ++i) {
            int r = i + ((lane >> 4) << 3);
            int c = kt * 16 + (lane & 15);
            sS[r * 1024 + c] = acc[i] * 0.125f;   // 1/sqrt(64)
        }
    }
    __syncthreads();

    // ---- softmax per row; emit f32 weights + bf16 probs in fragment layout ----
    {
        const int r   = t >> 3;      // 0..15
        const int seg = t & 7;       // 128 cols each
        const int c0  = seg * 128;

        float mx = -3.4e38f;
        for (int c = c0; c < c0 + 128; ++c) mx = fmaxf(mx, sS[r * 1024 + c]);
        sR[r * 8 + seg] = mx;
        __syncthreads();
        float rmx = sR[r * 8];
        #pragma unroll
        for (int j = 1; j < 8; ++j) rmx = fmaxf(rmx, sR[r * 8 + j]);

        float sum = 0.f;
        for (int c = c0; c < c0 + 128; ++c) {
            float e = __expf(sS[r * 1024 + c] - rmx);
            sS[r * 1024 + c] = e;
            sum += e;
        }
        sR[128 + r * 8 + seg] = sum;
        __syncthreads();
        float rs = 0.f;
        #pragma unroll
        for (int j = 0; j < 8; ++j) rs += sR[128 + r * 8 + j];
        float inv = 1.f / rs;

        float* outRow = attnW + (bh * SEQ + qt * 16 + r) * (size_t)SEQ;
        for (int c = c0; c < c0 + 128; ++c) {
            float p = sS[r * 1024 + c] * inv;
            outRow[c] = p;                               // f32 attention weights
            int hi, j; kswz(c & 31, hi, j);
            sP[(c >> 5) * 512 + ((((hi << 4) | r) << 4) | j)] = f2bf(p);
        }
        __syncthreads();
    }

    // ---- ctx = P @ V ; wave w owns d-range [w*16, w*16+16) ----
    f32x8 cacc = {0.f,0.f,0.f,0.f,0.f,0.f,0.f,0.f};
    for (int kc = 0; kc < 32; ++kc) {
        bf16x16 pa = *(const bf16x16*)(sP + kc * 512 + lane * 16);            // LDS b128
        bf16x16 vb = *(const bf16x16*)(Vf + ((bh * 32 + kc) * 4 + w) * 512 + lane * 16);
        cacc = wmma_bf16(pa, vb, cacc);
    }

    // scatter ctx directly into the final GEMM's A-fragment layout
    #pragma unroll
    for (int i = 0; i < 8; ++i) {
        int r = i + ((lane >> 4) << 3);
        int m = b * SEQ + qt * 16 + r;
        int col = h * D_K + w * 16 + (lane & 15);
        Cf[afrag_idx(m, col)] = f2bf(cacc[i]);
    }
}

// ---------- host launch ----------

extern "C" void kernel_launch(void* const* d_in, const int* in_sizes, int n_in,
                              void* d_out, int out_size, void* d_ws, size_t ws_size,
                              hipStream_t stream) {
    (void)in_sizes; (void)n_in; (void)out_size; (void)ws_size;

    const float* x  = (const float*)d_in[0];
    const float* Wq = (const float*)d_in[1];
    const float* bq = (const float*)d_in[2];
    const float* Wk = (const float*)d_in[3];
    const float* bk = (const float*)d_in[4];
    const float* Wv = (const float*)d_in[5];
    const float* bv = (const float*)d_in[6];
    const float* Wo = (const float*)d_in[7];
    const float* bo = (const float*)d_in[8];

    float* out  = (float*)d_out;                         // (B,S,512) f32
    float* attn = out + (size_t)M_TOTAL * D_MODEL;       // (B,H,S,S) f32

    char* ws = (char*)d_ws;
    unsigned short* xbf = (unsigned short*)ws; ws += (size_t)M_TOTAL * 512 * 2;
    unsigned short* wqb = (unsigned short*)ws; ws += 512 * 512 * 2;
    unsigned short* wkb = (unsigned short*)ws; ws += 512 * 512 * 2;
    unsigned short* wvb = (unsigned short*)ws; ws += 512 * 512 * 2;
    unsigned short* wob = (unsigned short*)ws; ws += 512 * 512 * 2;
    unsigned short* Qb  = (unsigned short*)ws; ws += (size_t)M_TOTAL * 512 * 2;
    unsigned short* Kb  = (unsigned short*)ws; ws += (size_t)M_TOTAL * 512 * 2;
    unsigned short* Vb  = (unsigned short*)ws; ws += (size_t)M_TOTAL * 512 * 2;
    unsigned short* Cb  = (unsigned short*)ws; ws += (size_t)M_TOTAL * 512 * 2;

    const int nX = M_TOTAL * 512;
    const int nW = 512 * 512;
    k_conv_A<<<(nX + 255) / 256, 256, 0, stream>>>(x, xbf, nX);
    k_conv_W<<<(nW + 255) / 256, 256, 0, stream>>>(Wq, wqb);
    k_conv_W<<<(nW + 255) / 256, 256, 0, stream>>>(Wk, wkb);
    k_conv_W<<<(nW + 255) / 256, 256, 0, stream>>>(Wv, wvb);
    k_conv_W<<<(nW + 255) / 256, 256, 0, stream>>>(Wo, wob);

    dim3 gg(M_TOTAL / 128, 512 / 64);
    k_gemm512<<<gg, 256, 0, stream>>>(xbf, wqb, bq, nullptr, Qb, 1);
    k_gemm512<<<gg, 256, 0, stream>>>(xbf, wkb, bk, nullptr, Kb, 1);
    k_gemm512<<<gg, 256, 0, stream>>>(xbf, wvb, bv, nullptr, Vb, 2);

    dim3 ga(SEQ / 16, N_HEADS, BATCH);
    size_t smem = 65536 + 32768 + 1024;   // 99,328 B (WGP LDS = 320 KB)
    k_attention<<<ga, 128, smem, stream>>>(Qb, Kb, Vb, attn, Cb);

    k_gemm512<<<gg, 256, 0, stream>>>(Cb, wob, bo, out, nullptr, 0);
}
